// Matcher_42305427866178
// MI455X (gfx1250) — matrix operand order repaired and checked by reading, not compiled, not run on hardware
//
#include <hip/hip_runtime.h>
#include <cstddef>
#include <cstdint>

typedef __attribute__((ext_vector_type(16))) _Float16 v16h;
typedef __attribute__((ext_vector_type(8)))  float    v8f;

#define NB 4
#define LL 3600
#define CCH 256
#define MM 2000

// ---------------- WMMA helpers (gfx1250 wave32 layouts) ----------------
__device__ __forceinline__ v8f wmma_f16(v16h a, v16h b, v8f c) {
  return __builtin_amdgcn_wmma_f32_16x16x32_f16(false, a, false, b, (short)0, c, false, false);
}

// 16-bit A 16x32 (MxK): lane g=lane>>4, M=lane&15; VGPR v holds K pair
__device__ __forceinline__ v16h frag_a16(const _Float16* As, int sk, int lane) {
  int g = lane >> 4; int mrow = lane & 15;
  const _Float16* row = As + mrow * sk;
  v16h a;
#pragma unroll
  for (int v = 0; v < 8; ++v) {
    int k = 2 * v + 16 * g;
    a[2 * v]     = row[k];
    a[2 * v + 1] = row[k + 1];
  }
  return a;
}

// 16-bit B 32x16 (KxN) from LDS stored [K][N] (row-major KxN, stride sn)
__device__ __forceinline__ v16h frag_b_nn16(const _Float16* Bs, int sn, int nOff, int lane) {
  int g = lane >> 4; int n = (lane & 15) + nOff;
  v16h b;
#pragma unroll
  for (int h = 0; h < 16; ++h) b[h] = Bs[(h + 16 * g) * sn + n];
  return b;
}

// 16-bit B 32x16 (KxN) from LDS stored [N][K] (i.e. B^T, stride sk) -> A@B^T
__device__ __forceinline__ v16h frag_b_nt16(const _Float16* Bs, int sk, int nOff, int lane) {
  int g = lane >> 4; int n = (lane & 15) + nOff;
  const _Float16* row = Bs + n * sk;
  v16h b;
#pragma unroll
  for (int h = 0; h < 16; ++h) b[h] = row[h + 16 * g];
  return b;
}

// ---------------- CDNA5 async global->LDS staging ----------------
// One lane moves 16B directly into LDS; tracked by ASYNCcnt.
__device__ __forceinline__ void async_lds_b128(const void* gaddr, unsigned lds_off) {
  asm volatile("global_load_async_to_lds_b128 %0, %1, off"
               :: "v"(lds_off), "v"(gaddr)
               : "memory");
}
__device__ __forceinline__ void wait_async_all() {
  asm volatile("s_wait_asynccnt 0x0" ::: "memory");
}
__device__ __forceinline__ unsigned lds_addr_of(const void* p) {
  // generic LDS pointer: low 32 bits are the LDS byte address (aperture rule)
  return (unsigned)(uintptr_t)p;
}

__device__ __forceinline__ float gelu_tanh(float x) {
  float x3 = x * x * x;
  return 0.5f * x * (1.f + tanhf(0.7978845608028654f * (x + 0.044715f * x3)));
}

// ---------------- elementwise utilities ----------------
__global__ void k_zero(float* p, long n) {
  long i = blockIdx.x * (long)blockDim.x + threadIdx.x;
  if (i < n) p[i] = 0.f;
}

__global__ void k_f32_to_f16(const float* x, _Float16* y, long n, float s) {
  long i = blockIdx.x * (long)blockDim.x + threadIdx.x;
  if (i < n) y[i] = (_Float16)(x[i] * s);
}

__global__ void k_pack_w16(const float* src, _Float16* dst, int rows, int cols, int dstStride) {
  int i = blockIdx.x * blockDim.x + threadIdx.x;
  if (i >= rows * cols) return;
  int r = i / cols, c = i % cols;
  dst[r * dstStride + c] = (_Float16)src[i];
}

// ---------------- coarse stage: WMMA GEMM tiles + exp row/col sums ----------------
// double-buffered async LDS staging; boundary rows handled by address clamping
// (clamped rows/cols only feed output elements the epilogue discards)
__global__ void k_coarse(const _Float16* A, const _Float16* B,
                         const unsigned char* m0, const unsigned char* m1,
                         float* rowsum, float* colsum, int L, int K) {
  __shared__ _Float16 As[2][64 * 32];
  __shared__ _Float16 Bs[2][64 * 32];
  __shared__ float Es[4][16 * 64];
  int n = blockIdx.z;
  int lBase = blockIdx.x * 64, sBase = blockIdx.y * 64;
  int tid = threadIdx.x, wave = tid >> 5, lane = tid & 31;
  const _Float16* Ab = A + (size_t)n * L * K;
  const _Float16* Bb = B + (size_t)n * L * K;
  v8f acc[4];
#pragma unroll
  for (int t = 0; t < 4; ++t)
#pragma unroll
    for (int r = 0; r < 8; ++r) acc[t][r] = 0.f;

  auto load_tiles = [&](int buf, int k0) {
#pragma unroll
    for (int j = 0; j < 2; ++j) {
      int seg = tid + 128 * j;            // 256 x 16B segments per tile
      int r = seg >> 2, s4 = seg & 3;     // row, 16B chunk within 64B row
      int l = lBase + r; if (l > L - 1) l = L - 1;
      int s = sBase + r; if (s > L - 1) s = L - 1;
      async_lds_b128(Ab + (size_t)l * K + k0 + s4 * 8,
                     lds_addr_of(&As[buf][r * 32 + s4 * 8]));
      async_lds_b128(Bb + (size_t)s * K + k0 + s4 * 8,
                     lds_addr_of(&Bs[buf][r * 32 + s4 * 8]));
    }
  };

  int p = 0;
  load_tiles(0, 0);
  wait_async_all();
  __syncthreads();
  for (int k0 = 0; k0 < K; k0 += 32) {
    if (k0 + 32 < K) load_tiles(p ^ 1, k0 + 32);
    v16h af = frag_a16(&As[p][wave * 16 * 32], 32, lane);
#pragma unroll
    for (int t = 0; t < 4; ++t) {
      v16h bf = frag_b_nt16(&Bs[p][0], 32, t * 16, lane);
      acc[t] = wmma_f16(af, bf, acc[t]);
    }
    wait_async_all();
    __syncthreads();
    p ^= 1;
  }

  int g = lane >> 4, c0 = lane & 15;
#pragma unroll
  for (int t = 0; t < 4; ++t)
#pragma unroll
    for (int r = 0; r < 8; ++r) {
      int rr = r + 8 * g;
      int l = lBase + wave * 16 + rr;
      int s = sBase + t * 16 + c0;
      float e = 0.f;
      if (l < L && s < L && m0[(size_t)n * L + l] && m1[(size_t)n * L + s]) e = __expf(acc[t][r]);
      Es[wave][rr * 64 + t * 16 + c0] = e;
    }
  __syncthreads();
  if (lane < 16) {
    int l = lBase + wave * 16 + lane;
    if (l < L) {
      float s = 0.f;
      for (int c = 0; c < 64; ++c) s += Es[wave][lane * 64 + c];
      atomicAdd(rowsum + (size_t)n * L + l, s);
    }
  }
  for (int c = lane; c < 64; c += 32) {
    int s = sBase + c;
    if (s < L) {
      float sum = 0.f;
      for (int r = 0; r < 16; ++r) sum += Es[wave][r * 64 + c];
      atomicAdd(colsum + (size_t)n * L + s, sum);
    }
  }
}

// focal loss at positive positions only (recompute the one dot product needed)
__global__ void k_pos(const int* conf, const _Float16* A, const _Float16* B,
                      const unsigned char* m0, const unsigned char* m1,
                      const float* rowsum, const float* colsum,
                      float* accs, int L, int K, long total) {
  long idx = blockIdx.x * (long)blockDim.x + threadIdx.x;
  if (idx >= total) return;
  if (conf[idx] != 1) return;
  int s = (int)(idx % L);
  long t = idx / L;
  int l = (int)(t % L);
  int n = (int)(t / L);
  atomicAdd(accs + 1, 1.f);
  if (!(m0[(size_t)n * L + l] && m1[(size_t)n * L + s])) return;
  const _Float16* pa = A + ((size_t)n * L + l) * K;
  const _Float16* pb = B + ((size_t)n * L + s) * K;
  __builtin_prefetch(pa, 0, 3);
  __builtin_prefetch(pb, 0, 3);
  float sim = 0.f;
  for (int d = 0; d < K; ++d) sim += (float)pa[d] * (float)pb[d];
  float e = __expf(sim);
  float cf = (e / rowsum[(size_t)n * L + l]) * (e / colsum[(size_t)n * L + s]);
  cf = fminf(fmaxf(cf, 1e-6f), 1.f - 1e-6f);
  float om = 1.f - cf;
  atomicAdd(accs + 0, -0.25f * om * om * __logf(cf));
}

// ---------------- fine stage ----------------
__global__ void k_gather(const float* fmap, const int* b_ids, const float* coords,
                         float* out, int Kwin, int Wsz, long total) {
  long idx = blockIdx.x * (long)blockDim.x + threadIdx.x;
  if (idx >= total) return;
  int c = (int)(idx & 63);
  long mk = idx >> 6;
  int k = (int)(mk % Kwin);
  int m = (int)(mk / Kwin);
  int half = Wsz >> 1;
  float x = coords[2 * m] + (float)(k % Wsz - half);
  float y = coords[2 * m + 1] + (float)(k / Wsz - half);
  int b = b_ids[m];
  float x0 = floorf(x), y0 = floorf(y);
  float wx = x - x0, wy = y - y0;
  const float* base = fmap + ((size_t)b * 64 + c) * (240 * 240);
  float v = 0.f;
#pragma unroll
  for (int dy = 0; dy < 2; ++dy)
#pragma unroll
    for (int dx = 0; dx < 2; ++dx) {
      float xi = x0 + dx, yi = y0 + dy;
      if (xi >= 0.f && xi <= 239.f && yi >= 0.f && yi <= 239.f) {
        float wgt = (dx ? wx : 1.f - wx) * (dy ? wy : 1.f - wy);
        v += base[(int)yi * 240 + (int)xi] * wgt;
      }
    }
  out[idx] = v;
}

// LayerNorm over 64 dims, one wave per row, f16 output
__global__ void k_ln64(const float* x, _Float16* y, int rows) {
  int row = blockIdx.x * 8 + (threadIdx.x >> 5);
  int lane = threadIdx.x & 31;
  if (row >= rows) return;
  const float* p = x + (size_t)row * 64;
  float a = p[lane], b = p[lane + 32];
  float s = a + b;
#pragma unroll
  for (int m = 16; m; m >>= 1) s += __shfl_xor(s, m, 32);
  float mean = s * (1.f / 64.f);
  float da = a - mean, db = b - mean;
  float q = da * da + db * db;
#pragma unroll
  for (int m = 16; m; m >>= 1) q += __shfl_xor(q, m, 32);
  float rstd = rsqrtf(q * (1.f / 64.f) + 1e-5f);
  _Float16* o = y + (size_t)row * 64;
  o[lane] = (_Float16)(da * rstd);
  o[lane + 32] = (_Float16)(db * rstd);
}

// Generic f16 WMMA GEMM: out = act(A[M,K] @ B[K,N] + bias) + res ; f32 and/or f16 stores
// double-buffered async LDS staging (N and K multiples of 8/32 in all uses)
__global__ void k_gemm(const _Float16* A, const _Float16* B, int M, int N, int K,
                       const float* bias, const float* res, float* outF, _Float16* outH, int act) {
  __shared__ _Float16 As[2][64 * 32];
  __shared__ _Float16 Bs[2][32 * 64];
  int tid = threadIdx.x, wave = tid >> 5, lane = tid & 31;
  int mBase = blockIdx.x * 64, nBase = blockIdx.y * 64;
  int mSub = wave & 3, nHalf = wave >> 2;
  v8f acc[2];
#pragma unroll
  for (int t = 0; t < 2; ++t)
#pragma unroll
    for (int r = 0; r < 8; ++r) acc[t][r] = 0.f;

  auto load_tiles = [&](int buf, int k0) {
    // A tile: 256 x 16B segments, 4 per 64B row
    {
      int r = tid >> 2, s4 = tid & 3;
      int m = mBase + r; if (m > M - 1) m = M - 1;
      async_lds_b128(A + (size_t)m * K + k0 + s4 * 8,
                     lds_addr_of(&As[buf][r * 32 + s4 * 8]));
    }
    // B tile [32][64]: 256 x 16B segments, 8 per 128B row
    {
      int r = tid >> 3, s8 = tid & 7;
      int col = nBase + s8 * 8; if (col > N - 8) col = N - 8;
      async_lds_b128(B + (size_t)(k0 + r) * N + col,
                     lds_addr_of(&Bs[buf][r * 64 + s8 * 8]));
    }
  };

  int p = 0;
  load_tiles(0, 0);
  wait_async_all();
  __syncthreads();
  for (int k0 = 0; k0 < K; k0 += 32) {
    if (k0 + 32 < K) load_tiles(p ^ 1, k0 + 32);
    v16h af = frag_a16(&As[p][mSub * 16 * 32], 32, lane);
#pragma unroll
    for (int t = 0; t < 2; ++t) {
      v16h bf = frag_b_nn16(&Bs[p][0], 64, nHalf * 32 + t * 16, lane);
      acc[t] = wmma_f16(af, bf, acc[t]);
    }
    wait_async_all();
    __syncthreads();
    p ^= 1;
  }

  int g = lane >> 4, c0 = lane & 15;
#pragma unroll
  for (int t = 0; t < 2; ++t)
#pragma unroll
    for (int r = 0; r < 8; ++r) {
      int m = mBase + mSub * 16 + r + 8 * g;
      int col = nBase + nHalf * 32 + t * 16 + c0;
      if (m < M && col < N) {
        float v = acc[t][r];
        if (bias) v += bias[col];
        if (act) v = gelu_tanh(v);
        if (res) v += res[(size_t)m * N + col];
        if (outF) outF[(size_t)m * N + col] = v;
        if (outH) outH[(size_t)m * N + col] = (_Float16)v;
      }
    }
}

// per-match attention core: softmax(q k^T / sqrt(32)) v @ Wo, residual into t
__global__ void k_attn(const _Float16* Pq, const _Float16* Pkv, float* t,
                       const float* Wo, int T) {
  __shared__ float q[49 * 32], kk[49 * 32], vv[49 * 32], sc[49 * 49], o[49 * 32];
  int m = blockIdx.x, tid = threadIdx.x;
  for (int i = tid; i < T * 32; i += 128) {
    int r = i >> 5, d = i & 31;
    size_t rowBase = (size_t)(m * T + r) * 96;
    q[i]  = (float)Pq[rowBase + d];
    kk[i] = (float)Pkv[rowBase + 32 + d];
    vv[i] = (float)Pkv[rowBase + 64 + d];
  }
  __syncthreads();
  for (int i = tid; i < T * T; i += 128) {
    int r = i / T, c = i % T;
    float s = 0.f;
    for (int d = 0; d < 32; ++d) s += q[r * 32 + d] * kk[c * 32 + d];
    sc[i] = s * 0.17677669529663687f;
  }
  __syncthreads();
  for (int r = tid; r < T; r += 128) {
    float mx = -1e30f;
    for (int j = 0; j < T; ++j) mx = fmaxf(mx, sc[r * T + j]);
    float sum = 0.f;
    for (int j = 0; j < T; ++j) { float e = __expf(sc[r * T + j] - mx); sc[r * T + j] = e; sum += e; }
    float inv = 1.f / sum;
    for (int j = 0; j < T; ++j) sc[r * T + j] *= inv;
  }
  __syncthreads();
  for (int i = tid; i < T * 32; i += 128) {
    int r = i >> 5, d = i & 31;
    float s = 0.f;
    for (int j = 0; j < T; ++j) s += sc[r * T + j] * vv[j * 32 + d];
    o[i] = s;
  }
  __syncthreads();
  for (int i = tid; i < T * 64; i += 128) {
    int r = i >> 6, c = i & 63;
    float s = 0.f;
    for (int d = 0; d < 32; ++d) s += o[r * 32 + d] * Wo[d * 64 + c];
    t[(size_t)(m * T + r) * 64 + c] += s;
  }
}

// regression heads + fine losses (one block per match)
template <int T_, int DIN_, int DH_, int MID_, bool CLS_>
__global__ void k_heads(const float* tA, const float* tB, const float* offs,
                        const float* W1, const float* b1, const float* gg, const float* bb,
                        const float* W2, const float* b2, float radius,
                        float* accSum, float* accCls, float* accCnt) {
  __shared__ float a[T_ * 64], bm[T_ * 64], smm[T_ * T_], tok[DIN_], h[DH_], red[256], red2[256];
  int m = blockIdx.x, tid = threadIdx.x;
  for (int i = tid; i < T_ * 64; i += 256) {
    a[i]  = tA[(size_t)m * T_ * 64 + i];
    bm[i] = tB[(size_t)m * T_ * 64 + i];
  }
  __syncthreads();
  if (CLS_) {
    for (int i = tid; i < T_ * T_; i += 256) {
      int r = i / T_, c = i % T_;
      float s = 0.f;
      for (int d = 0; d < 64; ++d) s += a[r * 64 + d] * bm[c * 64 + d];
      smm[i] = s * 0.125f;
    }
  } else {
    for (int i = tid; i < T_; i += 256) {
      float s = 0.f;
      for (int d = 0; d < 64; ++d) s += a[MID_ * 64 + d] * bm[i * 64 + d];
      smm[i] = s * 0.125f;
    }
  }
  __syncthreads();
  for (int i = tid; i < DIN_; i += 256) {
    float v;
    if (i < 64) v = a[MID_ * 64 + i];
    else if (i < 128) v = bm[MID_ * 64 + (i - 64)];
    else v = CLS_ ? smm[MID_ * T_ + (i - 128)] : smm[i - 128];
    tok[i] = v;
  }
  __syncthreads();
  for (int j = tid; j < DH_; j += 256) {
    float s = b1[j];
    for (int i = 0; i < DIN_; ++i) s += tok[i] * W1[i * DH_ + j];
    h[j] = s;
  }
  __syncthreads();
  float ps = 0.f, pq = 0.f;
  for (int j = tid; j < DH_; j += 256) { ps += h[j]; pq += h[j] * h[j]; }
  red[tid] = ps; red2[tid] = pq; __syncthreads();
  for (int s = 128; s; s >>= 1) {
    if (tid < s) { red[tid] += red[tid + s]; red2[tid] += red2[tid + s]; }
    __syncthreads();
  }
  float mean = red[0] * (1.f / DH_);
  float var = fmaxf(red2[0] * (1.f / DH_) - mean * mean, 0.f);
  float rstd = rsqrtf(var + 1e-5f);
  __syncthreads();
  for (int j = tid; j < DH_; j += 256) {
    float v = (h[j] - mean) * rstd * gg[j] + bb[j];
    h[j] = gelu_tanh(v);
  }
  __syncthreads();
  float p0 = 0.f, p1 = 0.f;
  for (int j = tid; j < DH_; j += 256) { p0 += h[j] * W2[2 * j]; p1 += h[j] * W2[2 * j + 1]; }
  red[tid] = p0; red2[tid] = p1; __syncthreads();
  for (int s = 128; s; s >>= 1) {
    if (tid < s) { red[tid] += red[tid + s]; red2[tid] += red2[tid + s]; }
    __syncthreads();
  }
  if (tid == 0) {
    float pr0 = tanhf(red[0] + b2[0]);
    float pr1 = tanhf(red2[0] + b2[1]);
    float gx = offs[2 * m] / radius, gy = offs[2 * m + 1] / radius;
    float lo = (gx - pr0) * (gx - pr0) + (gy - pr1) * (gy - pr1);
    bool inb = (gx >= -1.f && gx <= 1.f && gy >= -1.f && gy <= 1.f);
    if (inb) {
      atomicAdd(accSum, lo);
      atomicAdd(accCnt, 1.f);
      if (CLS_) {
        float cx = fminf(fmaxf(rintf(offs[2 * m] + 3.f), 0.f), 6.f);
        float cy = fminf(fmaxf(rintf(offs[2 * m + 1] + 3.f), 0.f), 6.f);
        int gti = (int)cx + (int)cy * 7;
        float mxc = -1e30f;
        for (int l = 0; l < T_; ++l) mxc = fmaxf(mxc, smm[l * T_ + gti]);
        float dc = 0.f;
        for (int l = 0; l < T_; ++l) dc += __expf(smm[l * T_ + gti] - mxc);
        float mxr = -1e30f;
        for (int j = 0; j < T_; ++j) mxr = fmaxf(mxr, smm[MID_ * T_ + j]);
        float dr = 0.f;
        for (int j = 0; j < T_; ++j) dr += __expf(smm[MID_ * T_ + j] - mxr);
        float sel = (__expf(smm[MID_ * T_ + gti] - mxc) / dc) * (__expf(smm[MID_ * T_ + gti] - mxr) / dr);
        atomicAdd(accCls, -0.1f * (1.f - sel) * __logf(sel + 1e-6f));
      }
    }
  }
}

__global__ void k_finalize(const float* accs, float* out) {
  if (threadIdx.x == 0 && blockIdx.x == 0) {
    out[0] = accs[0] / fmaxf(accs[1], 1.f);
    float c1 = fmaxf(accs[4], 1.f), c2 = fmaxf(accs[6], 1.f);
    out[1] = accs[2] / c1 + accs[3] / c1 + accs[5] / c2;
  }
}

// ---------------- host ----------------
extern "C" void kernel_launch(void* const* d_in, const int* in_sizes, int n_in,
                              void* d_out, int out_size, void* d_ws, size_t ws_size,
                              hipStream_t stream) {
  (void)in_sizes; (void)n_in; (void)out_size; (void)ws_size;
  const float* fc0       = (const float*)d_in[0];
  const float* fc1       = (const float*)d_in[1];
  const float* f0d       = (const float*)d_in[2];
  const float* f1d       = (const float*)d_in[3];
  const int*   conf      = (const int*)d_in[4];
  const unsigned char* mask0 = (const unsigned char*)d_in[5];
  const unsigned char* mask1 = (const unsigned char*)d_in[6];
  const int*   b_ids     = (const int*)d_in[7];
  const float* coords0   = (const float*)d_in[8];
  const float* coords1   = (const float*)d_in[9];
  const float* coords1s2 = (const float*)d_in[10];
  const float* offsets   = (const float*)d_in[11];
  const float* offsets2  = (const float*)d_in[12];
  // params flattened alphabetically: attn{1,2}: W1 W2 Wk Wo Wq Wv b1 b2 ; reg{1,2}: W1 W2 b1 b2 beta g
  const float* a1W1 = (const float*)d_in[13];
  const float* a1W2 = (const float*)d_in[14];
  const float* a1Wk = (const float*)d_in[15];
  const float* a1Wo = (const float*)d_in[16];
  const float* a1Wq = (const float*)d_in[17];
  const float* a1Wv = (const float*)d_in[18];
  const float* a1b1 = (const float*)d_in[19];
  const float* a1b2 = (const float*)d_in[20];
  const float* a2W1 = (const float*)d_in[21];
  const float* a2W2 = (const float*)d_in[22];
  const float* a2Wk = (const float*)d_in[23];
  const float* a2Wo = (const float*)d_in[24];
  const float* a2Wq = (const float*)d_in[25];
  const float* a2Wv = (const float*)d_in[26];
  const float* a2b1 = (const float*)d_in[27];
  const float* a2b2 = (const float*)d_in[28];
  const float* r1W1 = (const float*)d_in[29];
  const float* r1W2 = (const float*)d_in[30];
  const float* r1b1 = (const float*)d_in[31];
  const float* r1b2 = (const float*)d_in[32];
  const float* r1be = (const float*)d_in[33];
  const float* r1g  = (const float*)d_in[34];
  const float* r2W1 = (const float*)d_in[35];
  const float* r2W2 = (const float*)d_in[36];
  const float* r2b1 = (const float*)d_in[37];
  const float* r2b2 = (const float*)d_in[38];
  const float* r2be = (const float*)d_in[39];
  const float* r2g  = (const float*)d_in[40];

  // workspace carve (deterministic every call)
  char* base = (char*)d_ws;
  size_t off = 0;
  auto carve = [&](size_t bytes) -> char* {
    char* p = base + off;
    off += (bytes + 255) & ~(size_t)255;
    return p;
  };
  float*    rowsum = (float*)carve((size_t)NB * LL * 4);
  float*    colsum = (float*)carve((size_t)NB * LL * 4);
  float*    accs   = (float*)carve(256);
  _Float16* a16    = (_Float16*)carve((size_t)NB * LL * CCH * 2);
  _Float16* b16    = (_Float16*)carve((size_t)NB * LL * CCH * 2);
  float*    t0     = (float*)carve((size_t)MM * 49 * 64 * 4);
  float*    t1     = (float*)carve((size_t)MM * 49 * 64 * 4);
  float*    t0s2   = (float*)carve((size_t)MM * 9 * 64 * 4);
  float*    t1s2   = (float*)carve((size_t)MM * 9 * 64 * 4);
  _Float16* A16    = (_Float16*)carve((size_t)MM * 49 * 64 * 2);
  _Float16* P0h    = (_Float16*)carve((size_t)MM * 49 * 96 * 2);
  _Float16* P1h    = (_Float16*)carve((size_t)MM * 49 * 96 * 2);
  _Float16* H16    = (_Float16*)carve((size_t)MM * 49 * 256 * 2);
  _Float16* qkv1h  = (_Float16*)carve(64 * 96 * 2);
  _Float16* qkv2h  = (_Float16*)carve(64 * 96 * 2);
  _Float16* w1h1   = (_Float16*)carve(64 * 256 * 2);
  _Float16* w1h2   = (_Float16*)carve(64 * 256 * 2);
  _Float16* w2h1   = (_Float16*)carve(256 * 64 * 2);
  _Float16* w2h2   = (_Float16*)carve(256 * 64 * 2);

  auto cdiv = [](long a, long b) -> unsigned { return (unsigned)((a + b - 1) / b); };

  // zero accumulators (rowsum, colsum, accs are contiguous)
  long nz = (long)NB * LL * 2 + 64;
  k_zero<<<cdiv(nz, 256), 256, 0, stream>>>(rowsum, nz);

  // ---- coarse ----
  long nf = (long)NB * LL * CCH;
  k_f32_to_f16<<<cdiv(nf, 256), 256, 0, stream>>>(fc0, a16, nf, 1.f / 16.f);
  k_f32_to_f16<<<cdiv(nf, 256), 256, 0, stream>>>(fc1, b16, nf, 0.625f);  // (1/16)/0.1
  dim3 gC(cdiv(LL, 64), cdiv(LL, 64), NB);
  k_coarse<<<gC, 128, 0, stream>>>(a16, b16, mask0, mask1, rowsum, colsum, LL, CCH);
  long total = (long)NB * LL * LL;
  k_pos<<<cdiv(total, 256), 256, 0, stream>>>(conf, a16, b16, mask0, mask1,
                                              rowsum, colsum, accs, LL, CCH, total);

  // ---- fine: gathers ----
  long g1 = (long)MM * 49 * 64, g2 = (long)MM * 9 * 64;
  k_gather<<<cdiv(g1, 256), 256, 0, stream>>>(f0d, b_ids, coords0, t0, 49, 7, g1);
  k_gather<<<cdiv(g1, 256), 256, 0, stream>>>(f1d, b_ids, coords1, t1, 49, 7, g1);
  k_gather<<<cdiv(g2, 256), 256, 0, stream>>>(f0d, b_ids, coords0, t0s2, 9, 3, g2);
  k_gather<<<cdiv(g2, 256), 256, 0, stream>>>(f1d, b_ids, coords1s2, t1s2, 9, 3, g2);

  // pack weights to f16 (Wq|Wk|Wv concatenated -> [64,96])
  k_pack_w16<<<cdiv(2048, 256), 256, 0, stream>>>(a1Wq, qkv1h + 0, 64, 32, 96);
  k_pack_w16<<<cdiv(2048, 256), 256, 0, stream>>>(a1Wk, qkv1h + 32, 64, 32, 96);
  k_pack_w16<<<cdiv(2048, 256), 256, 0, stream>>>(a1Wv, qkv1h + 64, 64, 32, 96);
  k_pack_w16<<<cdiv(2048, 256), 256, 0, stream>>>(a2Wq, qkv2h + 0, 64, 32, 96);
  k_pack_w16<<<cdiv(2048, 256), 256, 0, stream>>>(a2Wk, qkv2h + 32, 64, 32, 96);
  k_pack_w16<<<cdiv(2048, 256), 256, 0, stream>>>(a2Wv, qkv2h + 64, 64, 32, 96);
  k_pack_w16<<<cdiv(16384, 256), 256, 0, stream>>>(a1W1, w1h1, 64, 256, 256);
  k_pack_w16<<<cdiv(16384, 256), 256, 0, stream>>>(a2W1, w1h2, 64, 256, 256);
  k_pack_w16<<<cdiv(16384, 256), 256, 0, stream>>>(a1W2, w2h1, 256, 64, 64);
  k_pack_w16<<<cdiv(16384, 256), 256, 0, stream>>>(a2W2, w2h2, 256, 64, 64);

  // ---- attention + MLP block (shared weights, both directions) ----
  auto run_block = [&](float* tA, float* tB, int T, int Ntok,
                       _Float16* qkvH, const float* Wo,
                       _Float16* w1h, const float* b1v, _Float16* w2h, const float* b2v) {
    unsigned gLn = cdiv(Ntok, 8);
    unsigned gM = cdiv(Ntok, 64);
    k_ln64<<<gLn, 256, 0, stream>>>(tA, A16, Ntok);
    k_gemm<<<dim3(gM, 2), 256, 0, stream>>>(A16, qkvH, Ntok, 96, 64, nullptr, nullptr, nullptr, P0h, 0);
    k_ln64<<<gLn, 256, 0, stream>>>(tB, A16, Ntok);
    k_gemm<<<dim3(gM, 2), 256, 0, stream>>>(A16, qkvH, Ntok, 96, 64, nullptr, nullptr, nullptr, P1h, 0);
    k_attn<<<MM, 128, 0, stream>>>(P0h, P1h, tA, Wo, T);
    k_attn<<<MM, 128, 0, stream>>>(P1h, P0h, tB, Wo, T);
    // MLP: x + gelu(ln(x)@W1+b1)@W2 + b2
    k_ln64<<<gLn, 256, 0, stream>>>(tA, A16, Ntok);
    k_gemm<<<dim3(gM, 4), 256, 0, stream>>>(A16, w1h, Ntok, 256, 64, b1v, nullptr, nullptr, H16, 1);
    k_gemm<<<dim3(gM, 1), 256, 0, stream>>>(H16, w2h, Ntok, 64, 256, b2v, tA, tA, nullptr, 0);
    k_ln64<<<gLn, 256, 0, stream>>>(tB, A16, Ntok);
    k_gemm<<<dim3(gM, 4), 256, 0, stream>>>(A16, w1h, Ntok, 256, 64, b1v, nullptr, nullptr, H16, 1);
    k_gemm<<<dim3(gM, 1), 256, 0, stream>>>(H16, w2h, Ntok, 64, 256, b2v, tB, tB, nullptr, 0);
  };
  run_block(t0, t1, 49, MM * 49, qkv1h, a1Wo, w1h1, a1b1, w2h1, a1b2);
  run_block(t0s2, t1s2, 9, MM * 9, qkv2h, a2Wo, w1h2, a2b1, w2h2, a2b2);

  // ---- heads + losses ----
  k_heads<49, 177, 354, 24, true><<<MM, 256, 0, stream>>>(
      t0, t1, offsets, r1W1, r1b1, r1g, r1be, r1W2, r1b2, 3.5f,
      accs + 2, accs + 3, accs + 4);
  k_heads<9, 137, 274, 4, false><<<MM, 256, 0, stream>>>(
      t0s2, t1s2, offsets2, r2W1, r2b1, r2g, r2be, r2W2, r2b2, 1.5f,
      accs + 5, accs + 5, accs + 6);

  k_finalize<<<1, 32, 0, stream>>>(accs, (float*)d_out);
}
